// ResidualAttentionBlock_11991548690553
// MI455X (gfx1250) — matrix operand order repaired and checked by reading, not compiled, not run on hardware
//
#include <hip/hip_runtime.h>
#include <math.h>

// ---------------- types ----------------
typedef unsigned short u16;
typedef unsigned int   u32;
typedef __attribute__((ext_vector_type(16))) __bf16 v16bf;
typedef __attribute__((ext_vector_type(8)))  float  v8f;
typedef __attribute__((ext_vector_type(4)))  u32    v4u;

union Frag { v16bf v; v4u q[2]; u16 s[16]; };

// float -> bf16 (round to nearest even)
__device__ inline u16 f2bf(float x) {
  u32 u = __float_as_uint(x);
  if ((u & 0x7F800000u) == 0x7F800000u) return (u16)(u >> 16); // inf/nan passthrough
  u32 r = u + 0x7FFFu + ((u >> 16) & 1u);
  return (u16)(r >> 16);
}

// ---------------- WMMA fragment loads (bf16, 16x16x32) ----------------
// A fragment (16x32, row-major source, K-contiguous):
//   lane l: m = m0 + (l&15), h = l>>4; elems e0..7 -> k = k0 + h*8 + e,
//   e8..15 -> k = k0 + 16 + h*8 + (e-8).   (ISA 7.12.2, 16-bit A 16x32)
__device__ inline v16bf load_frag_a(const u16* A, int lda, int m0, int k0, int lane) {
  int m = m0 + (lane & 15), h = lane >> 4;
  const u16* p = A + (size_t)m * lda + k0 + h * 8;
  Frag f;
  f.q[0] = *(const v4u*)p;
  f.q[1] = *(const v4u*)(p + 16);
  return f.v;
}
// B fragment (32x16): lane l holds column n = n0 + (l&15); sel = l>>4:
// elems j=0..7 -> k = k0 + sel*8 + j, j=8..15 -> k = k0 + 16 + sel*8 + (j-8).
// Source is row-major [N,K] (i.e. B[k][n] = W[n][k]).
__device__ inline v16bf load_frag_b(const u16* B, int ldb, int n0, int k0, int lane) {
  int n = n0 + (lane & 15), s = lane >> 4;
  const u16* p = B + (size_t)n * ldb + k0 + s * 8;
  Frag f;
  f.q[0] = *(const v4u*)p;
  f.q[1] = *(const v4u*)(p + 16);
  return f.v;
}

__device__ inline v8f wmma_bf16(v16bf a, v16bf b, v8f c) {
  return __builtin_amdgcn_wmma_f32_16x16x32_bf16(false, a, false, b, (short)0, c, false, false);
}

// ---------------- GEMM core: C[m,n] = sum_k A[m,k] * W[n,k] ----------------
// 256-thread block = 8 waves (4x2). Block tile 256x128; wave tile 64x64
// (4x4 of 16x16 => 16 WMMAs per K-step vs 8 fragment loads: 32 FLOP/byte).
__device__ inline void gemm_core(const u16* __restrict__ A, const u16* __restrict__ Bw,
                                 int lda, int ldb, int K,
                                 int mbase, int nbase, int lane, v8f acc[4][4]) {
  v8f z = {};
  for (int i = 0; i < 4; i++)
    for (int j = 0; j < 4; j++) acc[i][j] = z;
  for (int k = 0; k < K; k += 32) {
    v16bf a0 = load_frag_a(A, lda, mbase,      k, lane);
    v16bf a1 = load_frag_a(A, lda, mbase + 16, k, lane);
    v16bf a2 = load_frag_a(A, lda, mbase + 32, k, lane);
    v16bf a3 = load_frag_a(A, lda, mbase + 48, k, lane);
    v16bf b0 = load_frag_b(Bw, ldb, nbase,      k, lane);
    v16bf b1 = load_frag_b(Bw, ldb, nbase + 16, k, lane);
    v16bf b2 = load_frag_b(Bw, ldb, nbase + 32, k, lane);
    v16bf b3 = load_frag_b(Bw, ldb, nbase + 48, k, lane);
    acc[0][0] = wmma_bf16(a0, b0, acc[0][0]);
    acc[0][1] = wmma_bf16(a0, b1, acc[0][1]);
    acc[0][2] = wmma_bf16(a0, b2, acc[0][2]);
    acc[0][3] = wmma_bf16(a0, b3, acc[0][3]);
    acc[1][0] = wmma_bf16(a1, b0, acc[1][0]);
    acc[1][1] = wmma_bf16(a1, b1, acc[1][1]);
    acc[1][2] = wmma_bf16(a1, b2, acc[1][2]);
    acc[1][3] = wmma_bf16(a1, b3, acc[1][3]);
    acc[2][0] = wmma_bf16(a2, b0, acc[2][0]);
    acc[2][1] = wmma_bf16(a2, b1, acc[2][1]);
    acc[2][2] = wmma_bf16(a2, b2, acc[2][2]);
    acc[2][3] = wmma_bf16(a2, b3, acc[2][3]);
    acc[3][0] = wmma_bf16(a3, b0, acc[3][0]);
    acc[3][1] = wmma_bf16(a3, b1, acc[3][1]);
    acc[3][2] = wmma_bf16(a3, b2, acc[3][2]);
    acc[3][3] = wmma_bf16(a3, b3, acc[3][3]);
  }
}

// ---------------- elementwise fp32 -> bf16 ----------------
__global__ __launch_bounds__(256) void k_convert(const float* __restrict__ in,
                                                 u16* __restrict__ out, int n) {
  int i = blockIdx.x * 256 + threadIdx.x;
  if (i < n) out[i] = f2bf(in[i]);
}

// ---------------- LayerNorm (W = 1024), one row per block ----------------
__global__ __launch_bounds__(256) void k_layernorm(const float* __restrict__ x,
                                                   const float* __restrict__ w,
                                                   const float* __restrict__ b,
                                                   u16* __restrict__ out) {
  __shared__ float red[256];
  int row = blockIdx.x, tid = threadIdx.x;
  const float* xr = x + (size_t)row * 1024;
  float v[4], s = 0.f, s2 = 0.f;
  for (int i = 0; i < 4; i++) {
    float t = xr[tid + i * 256];
    v[i] = t; s += t; s2 += t * t;
  }
  red[tid] = s; __syncthreads();
  for (int off = 128; off > 0; off >>= 1) {
    if (tid < off) red[tid] += red[tid + off];
    __syncthreads();
  }
  float mean = red[0] * (1.f / 1024.f);
  __syncthreads();
  red[tid] = s2; __syncthreads();
  for (int off = 128; off > 0; off >>= 1) {
    if (tid < off) red[tid] += red[tid + off];
    __syncthreads();
  }
  float var = red[0] * (1.f / 1024.f) - mean * mean;
  float rstd = rsqrtf(var + 1e-5f);
  for (int i = 0; i < 4; i++) {
    int c = tid + i * 256;
    out[(size_t)row * 1024 + c] = f2bf((v[i] - mean) * rstd * w[c] + b[c]);
  }
}

// ---------------- QKV GEMM: [4096,1024] x [3072,1024]^T, scatter per head ----
// qkv reshape is (B,T,H,3C): channel n -> head n/192; r=n%192: q(r<64) k(<128) v.
// Q,K stored [B,H,T,64]; V stored transposed [B,H,64,T] for contiguous PV frags.
__global__ __launch_bounds__(256) void k_gemm_qkv(const u16* __restrict__ A,
                                                  const u16* __restrict__ Bw,
                                                  const float* __restrict__ bias,
                                                  u16* __restrict__ Qb,
                                                  u16* __restrict__ Kb,
                                                  u16* __restrict__ Vt) {
  int lane = threadIdx.x & 31, wv = threadIdx.x >> 5;
  int mbase = blockIdx.x * 256 + (wv >> 1) * 64;
  int nbase = blockIdx.y * 128 + (wv & 1) * 64;
  v8f acc[4][4];
  gemm_core(A, Bw, 1024, 1024, 1024, mbase, nbase, lane, acc);
  int hf = lane >> 4, nl = lane & 15;
  for (int j = 0; j < 4; j++) {
    int n = nbase + j * 16 + nl;
    float bs = bias[n];
    int head = n / 192, rr = n % 192;
    for (int i = 0; i < 4; i++) {
      for (int r = 0; r < 8; r++) {
        int m = mbase + i * 16 + hf * 8 + r;
        int bb = m >> 11, t = m & 2047;
        u16 bv = f2bf(acc[i][j][r] + bs);
        size_t bh = (size_t)(bb * 16 + head);
        if (rr < 64)        Qb[(bh * 2048 + t) * 64 + rr] = bv;
        else if (rr < 128)  Kb[(bh * 2048 + t) * 64 + (rr - 64)] = bv;
        else                Vt[(bh * 64 + (rr - 128)) * 2048 + t] = bv;
      }
    }
  }
}

// ---------------- Flash attention ----------------
// grid (T/128, B*H); block 256 = 8 waves; wave owns 16 query rows.
// Per key chunk of 32: S=Q K^T (4 wmma) -> LDS -> online softmax (scale 1/8)
// -> P bf16 in LDS -> O = O*corr + P V (4 wmma).
__global__ __launch_bounds__(256) void k_attention(const u16* __restrict__ Qb,
                                                   const u16* __restrict__ Kb,
                                                   const u16* __restrict__ Vt,
                                                   u16* __restrict__ Ob) {
  __shared__ __align__(16) float S[8][16 * 32];
  __shared__ __align__(16) u16   P[8][16 * 32];
  __shared__ float CORR[8][16];
  __shared__ float ROWL[8][16];
  int lane = threadIdx.x & 31, wv = threadIdx.x >> 5;
  int bh = blockIdx.y;
  int bb = bh >> 4, h = bh & 15;
  const u16* Q = Qb + (size_t)bh * 2048 * 64;
  const u16* K = Kb + (size_t)bh * 2048 * 64;
  const u16* V = Vt + (size_t)bh * 64 * 2048;
  int q0 = blockIdx.x * 128 + wv * 16;
  int nl = lane & 15, hf = lane >> 4;

  v16bf aq0 = load_frag_a(Q, 64, q0, 0, lane);
  v16bf aq1 = load_frag_a(Q, 64, q0, 32, lane);
  v8f z = {};
  v8f o0 = z, o1 = z, o2 = z, o3 = z;
  float mrow = -3.0e38f, lrow = 0.f;   // live in lanes 0..15

  for (int kb = 0; kb < 2048; kb += 32) {
    v8f s0 = z, s1 = z;
    s0 = wmma_bf16(aq0, load_frag_b(K, 64, kb,      0,  lane), s0);
    s0 = wmma_bf16(aq1, load_frag_b(K, 64, kb,      32, lane), s0);
    s1 = wmma_bf16(aq0, load_frag_b(K, 64, kb + 16, 0,  lane), s1);
    s1 = wmma_bf16(aq1, load_frag_b(K, 64, kb + 16, 32, lane), s1);

    float* Sw = S[wv];
    for (int r = 0; r < 8; r++) {
      Sw[(r + 8 * hf) * 32 + nl]      = s0[r];
      Sw[(r + 8 * hf) * 32 + 16 + nl] = s1[r];
    }
    __syncthreads();

    if (lane < 16) {
      const float* Sr = &S[wv][lane * 32];
      float mx = mrow;
      for (int jj = 0; jj < 32; jj++) mx = fmaxf(mx, Sr[jj] * 0.125f);
      float corr = __expf(mrow - mx);
      float sum = 0.f;
      u16* Pr = &P[wv][lane * 32];
      for (int jj = 0; jj < 32; jj++) {
        float p = __expf(Sr[jj] * 0.125f - mx);
        sum += p;
        Pr[jj] = f2bf(p);
      }
      lrow = lrow * corr + sum;
      mrow = mx;
      CORR[wv][lane] = corr;
    }
    __syncthreads();

    float c[8];
    for (int r = 0; r < 8; r++) c[r] = CORR[wv][r + 8 * hf];
    for (int r = 0; r < 8; r++) { o0[r] *= c[r]; o1[r] *= c[r]; o2[r] *= c[r]; o3[r] *= c[r]; }
    v16bf pf = load_frag_a(&P[wv][0], 32, 0, 0, lane);
    o0 = wmma_bf16(pf, load_frag_b(V, 2048, 0,  kb, lane), o0);
    o1 = wmma_bf16(pf, load_frag_b(V, 2048, 16, kb, lane), o1);
    o2 = wmma_bf16(pf, load_frag_b(V, 2048, 32, kb, lane), o2);
    o3 = wmma_bf16(pf, load_frag_b(V, 2048, 48, kb, lane), o3);
    __syncthreads();
  }

  if (lane < 16) ROWL[wv][lane] = lrow;
  __syncthreads();
  for (int r = 0; r < 8; r++) {
    int row = r + 8 * hf;
    float inv = 1.f / ROWL[wv][row];
    size_t base = ((size_t)(bb * 2048 + q0 + row)) * 1024 + h * 64;
    Ob[base + nl]      = f2bf(o0[r] * inv);
    Ob[base + 16 + nl] = f2bf(o1[r] * inv);
    Ob[base + 32 + nl] = f2bf(o2[r] * inv);
    Ob[base + 48 + nl] = f2bf(o3[r] * inv);
  }
}

// ---------------- proj GEMM + bias + residual -> x1 (fp32) ----------------
__global__ __launch_bounds__(256) void k_gemm_proj(const u16* __restrict__ A,
                                                   const u16* __restrict__ Bw,
                                                   const float* __restrict__ bias,
                                                   const float* __restrict__ resid,
                                                   float* __restrict__ x1) {
  int lane = threadIdx.x & 31, wv = threadIdx.x >> 5;
  int mbase = blockIdx.x * 256 + (wv >> 1) * 64;
  int nbase = blockIdx.y * 128 + (wv & 1) * 64;
  v8f acc[4][4];
  gemm_core(A, Bw, 1024, 1024, 1024, mbase, nbase, lane, acc);
  int hf = lane >> 4, nl = lane & 15;
  for (int j = 0; j < 4; j++) {
    int n = nbase + j * 16 + nl;
    float bs = bias[n];
    for (int i = 0; i < 4; i++)
      for (int r = 0; r < 8; r++) {
        int m = mbase + i * 16 + hf * 8 + r;
        size_t idx = (size_t)m * 1024 + n;
        x1[idx] = acc[i][j][r] + bs + resid[idx];
      }
  }
}

// ---------------- FC GEMM + bias + exact GELU -> bf16 h ----------------
__global__ __launch_bounds__(256) void k_gemm_fc(const u16* __restrict__ A,
                                                 const u16* __restrict__ Bw,
                                                 const float* __restrict__ bias,
                                                 u16* __restrict__ hbuf) {
  int lane = threadIdx.x & 31, wv = threadIdx.x >> 5;
  int mbase = blockIdx.x * 256 + (wv >> 1) * 64;
  int nbase = blockIdx.y * 128 + (wv & 1) * 64;
  v8f acc[4][4];
  gemm_core(A, Bw, 1024, 1024, 1024, mbase, nbase, lane, acc);
  int hf = lane >> 4, nl = lane & 15;
  for (int j = 0; j < 4; j++) {
    int n = nbase + j * 16 + nl;
    float bs = bias[n];
    for (int i = 0; i < 4; i++)
      for (int r = 0; r < 8; r++) {
        int m = mbase + i * 16 + hf * 8 + r;
        float t = acc[i][j][r] + bs;
        float g = 0.5f * t * (1.f + erff(t * 0.70710678118654752f));
        hbuf[(size_t)m * 4096 + n] = f2bf(g);
      }
  }
}

// ---------------- FCProj GEMM + bias + residual -> out (fp32) ----------------
__global__ __launch_bounds__(256) void k_gemm_fcproj(const u16* __restrict__ A,
                                                     const u16* __restrict__ Bw,
                                                     const float* __restrict__ bias,
                                                     const float* __restrict__ x1,
                                                     float* __restrict__ out) {
  int lane = threadIdx.x & 31, wv = threadIdx.x >> 5;
  int mbase = blockIdx.x * 256 + (wv >> 1) * 64;
  int nbase = blockIdx.y * 128 + (wv & 1) * 64;
  v8f acc[4][4];
  gemm_core(A, Bw, 4096, 4096, 4096, mbase, nbase, lane, acc);
  int hf = lane >> 4, nl = lane & 15;
  for (int j = 0; j < 4; j++) {
    int n = nbase + j * 16 + nl;
    float bs = bias[n];
    for (int i = 0; i < 4; i++)
      for (int r = 0; r < 8; r++) {
        int m = mbase + i * 16 + hf * 8 + r;
        size_t idx = (size_t)m * 1024 + n;
        out[idx] = acc[i][j][r] + bs + x1[idx];
      }
  }
}

// ---------------- launch ----------------
extern "C" void kernel_launch(void* const* d_in, const int* in_sizes, int n_in,
                              void* d_out, int out_size, void* d_ws, size_t ws_size,
                              hipStream_t stream) {
  const float* x        = (const float*)d_in[0];
  const float* ln1_w    = (const float*)d_in[1];
  const float* ln1_b    = (const float*)d_in[2];
  const float* qkv_w    = (const float*)d_in[3];
  const float* qkv_b    = (const float*)d_in[4];
  const float* proj_w   = (const float*)d_in[5];
  const float* proj_b   = (const float*)d_in[6];
  const float* ln2_w    = (const float*)d_in[7];
  const float* ln2_b    = (const float*)d_in[8];
  const float* fc_w     = (const float*)d_in[9];
  const float* fc_b     = (const float*)d_in[10];
  const float* fcproj_w = (const float*)d_in[11];
  const float* fcproj_b = (const float*)d_in[12];
  float* out = (float*)d_out;

  const size_t MT = 4096;            // B*T
  char* ws = (char*)d_ws;
  auto carve = [&](size_t bytes) -> char* {
    char* p = ws;
    ws += (bytes + 255) & ~(size_t)255;
    return p;
  };
  u16* wQ   = (u16*)carve(3145728u * 2);          // qkv_w bf16 [3072,1024]
  u16* wP   = (u16*)carve(1048576u * 2);          // proj_w bf16 [1024,1024]
  u16* wF   = (u16*)carve(4194304u * 2);          // fc_w bf16 [4096,1024]
  u16* wFP  = (u16*)carve(4194304u * 2);          // fcproj_w bf16 [1024,4096]
  u16* xln1 = (u16*)carve(MT * 1024 * 2);
  u16* Qb   = (u16*)carve(MT * 1024 * 2);         // [B,H,T,64]
  u16* Kb   = (u16*)carve(MT * 1024 * 2);         // [B,H,T,64]
  u16* Vt   = (u16*)carve(MT * 1024 * 2);         // [B,H,64,T]
  u16* Ob   = (u16*)carve(MT * 1024 * 2);         // attn out bf16 [4096,1024]
  float* x1 = (float*)carve(MT * 1024 * 4);       // residual-1 fp32
  u16* xln2 = (u16*)carve(MT * 1024 * 2);
  u16* hbuf = (u16*)carve(MT * 4096 * 2);         // gelu(fc) bf16 [4096,4096]

  // 1. weights -> bf16
  k_convert<<<(3145728 + 255) / 256, 256, 0, stream>>>(qkv_w, wQ, 3145728);
  k_convert<<<(1048576 + 255) / 256, 256, 0, stream>>>(proj_w, wP, 1048576);
  k_convert<<<(4194304 + 255) / 256, 256, 0, stream>>>(fc_w, wF, 4194304);
  k_convert<<<(4194304 + 255) / 256, 256, 0, stream>>>(fcproj_w, wFP, 4194304);

  // 2. LN1
  k_layernorm<<<4096, 256, 0, stream>>>(x, ln1_w, ln1_b, xln1);
  // 3. QKV GEMM (M=4096, N=3072), block tile 256x128
  k_gemm_qkv<<<dim3(16, 24), 256, 0, stream>>>(xln1, wQ, qkv_b, Qb, Kb, Vt);
  // 4. attention
  k_attention<<<dim3(16, 32), 256, 0, stream>>>(Qb, Kb, Vt, Ob);
  // 5. proj + residual (M=4096, N=1024)
  k_gemm_proj<<<dim3(16, 8), 256, 0, stream>>>(Ob, wP, proj_b, x, x1);
  // 6. LN2
  k_layernorm<<<4096, 256, 0, stream>>>(x1, ln2_w, ln2_b, xln2);
  // 7. FC + GELU (M=4096, N=4096)
  k_gemm_fc<<<dim3(16, 32), 256, 0, stream>>>(xln2, wF, fc_b, hbuf);
  // 8. FCProj + residual (M=4096, N=1024, K=4096)
  k_gemm_fcproj<<<dim3(16, 8), 256, 0, stream>>>(hbuf, wFP, fcproj_b, x1, out);
}
